// ANPMRShapeNet1D_23854248362556
// MI455X (gfx1250) — matrix operand order repaired and checked by reading, not compiled, not run on hardware
//
#include <hip/hip_runtime.h>
#include <math.h>

typedef __bf16 bf16;
typedef __attribute__((ext_vector_type(16))) __bf16 v16bf;
typedef __attribute__((ext_vector_type(8)))  __bf16 v8bf;
typedef __attribute__((ext_vector_type(8)))  float  v8f;
typedef int v4i_vs __attribute__((vector_size(16)));

// ---- model constants ----
#define TT     16
#define NQQ    32
#define DW     256
#define DR     256
#define DZ     128
#define NHEADS 8
#define NFEAT  1419   // int(256*ln(256))
#define NIMG   1024   // 512 ctx + 512 qry
#define TILE   32

// Use CDNA5 async global->LDS copies when the toolchain exposes them.
#if defined(__HIP_DEVICE_COMPILE__) && defined(__has_builtin)
#if __has_builtin(__builtin_amdgcn_global_load_async_to_lds_b128) && \
    __has_builtin(__builtin_amdgcn_s_wait_asynccnt)
#define ASYNC_LDS 1
#endif
#endif
#ifndef ASYNC_LDS
#define ASYNC_LDS 0
#endif

#if ASYNC_LDS
#define GV4(p) ((__attribute__((address_space(1))) v4i_vs*)(p))
#define LV4(p) ((__attribute__((address_space(3))) v4i_vs*)(p))
#endif

// =====================================================================
// Operand packing: fp32 (arbitrary strides) -> zero-padded bf16 tiles.
// A-pack: out[b][m][k], Mpad x Kpad.  B-pack: out[b][k][n], Kpad x Npad.
// =====================================================================
__global__ void pack_a_kernel(const float* __restrict__ in, bf16* __restrict__ out,
                              int M, int K, int Mpad, int Kpad,
                              long sAm, long sAk, long bsIn, int nb, float ascale)
{
    long idx = (long)blockIdx.x * blockDim.x + threadIdx.x;
    long per = (long)Mpad * Kpad;
    if (idx >= per * nb) return;
    int b = (int)(idx / per);
    long r = idx - (long)b * per;
    int m = (int)(r / Kpad), k = (int)(r % Kpad);
    float v = 0.f;
    if (m < M && k < K) v = in[(long)b*bsIn + (long)m*sAm + (long)k*sAk] * ascale;
    out[idx] = (bf16)v;
}

__global__ void pack_b_kernel(const float* __restrict__ in, bf16* __restrict__ out,
                              int N, int K, int Npad, int Kpad,
                              long sBn, long sBk, long bsIn, int nb)
{
    long idx = (long)blockIdx.x * blockDim.x + threadIdx.x;
    long per = (long)Kpad * Npad;
    if (idx >= per * nb) return;
    int b = (int)(idx / per);
    long r = idx - (long)b * per;
    int k = (int)(r / Npad), n = (int)(r % Npad);
    float v = 0.f;
    if (n < N && k < K) v = in[(long)b*bsIn + (long)n*sBn + (long)k*sBk];
    out[idx] = (bf16)v;
}

// =====================================================================
// Packed-bf16 GEMM: D[b][m][n] = act(A.B^T + bias). 128 threads = 4 waves,
// 32x32 output tile / block; each wave owns a 16x16 WMMA tile.
// Hot loop: b128 global->LDS staging (async CDNA5 copies when available),
// ds_load fragments, v_wmma_f32_16x16x32_bf16. No guards in the loop.
// =====================================================================
__global__ __launch_bounds__(128) void gemm_pk_wmma(
    const bf16* __restrict__ Ap, const bf16* __restrict__ Bp,
    const float* __restrict__ bias, float* __restrict__ D,
    int M, int N, int Kpad, int Npad,
    long bsA, long bsB, long bsBias, long bsD, int act)
{
    __shared__ bf16 As[TILE][TILE];   // [m][k]
    __shared__ bf16 Bs[TILE][TILE];   // [k][n]

    const int ntn = Npad >> 5;
    const int tm  = blockIdx.x / ntn;
    const int tn  = blockIdx.x % ntn;
    const int b   = blockIdx.y;
    Ap += (long)b * bsA;
    Bp += (long)b * bsB;
    D  += (long)b * bsD;
    const float* bb = bias ? (bias + (long)b * bsBias) : nullptr;

    const int t    = threadIdx.x;       // 0..127
    const int wave = t >> 5, lane = t & 31;
    const int wm   = wave >> 1, wn = wave & 1;
    const int half = lane >> 4, ml = lane & 15;
    const int srow = t >> 2, schk = t & 3;       // staging: 32 rows x 4 x 16B

    const bf16* gA = Ap + (long)(tm*TILE + srow) * Kpad + schk*8;
    const bf16* gB = Bp + (long)srow * Npad + tn*TILE + schk*8;

    v8f acc = {};
    for (int k0 = 0; k0 < Kpad; k0 += TILE) {
#if ASYNC_LDS
        __syncthreads();   // previous iteration's LDS reads done
        __builtin_amdgcn_global_load_async_to_lds_b128(
            GV4(gA + k0), LV4(&As[srow][schk*8]), 0, 0);
        __builtin_amdgcn_global_load_async_to_lds_b128(
            GV4(gB + (long)k0 * Npad), LV4(&Bs[srow][schk*8]), 0, 0);
        __builtin_amdgcn_s_wait_asynccnt(0);
        __syncthreads();
#else
        float4 ar = *(const float4*)(gA + k0);
        float4 br = *(const float4*)(gB + (long)k0 * Npad);
        __syncthreads();
        *(float4*)(&As[srow][schk*8]) = ar;
        *(float4*)(&Bs[srow][schk*8]) = br;
        __syncthreads();
#endif
        // A fragment (ISA 16-bit 16x32 layout): lane row = wm*16+ml,
        // K = {8*half+0..7} then {16+8*half+0..7} -> two contiguous runs.
        v8bf lo = *(const v8bf*)(&As[wm*16 + ml][8*half]);
        v8bf hi = *(const v8bf*)(&As[wm*16 + ml][16 + 8*half]);
        v16bf af;
        ((v8bf*)&af)[0] = lo;
        ((v8bf*)&af)[1] = hi;
        // B fragment: lane column = wn*16+ml, K in same order as A.
        v16bf bfm;
#pragma unroll
        for (int e = 0; e < 16; ++e)
            bfm[e] = Bs[(e < 8) ? (8*half + e) : (16 + 8*half + e - 8)][wn*16 + ml];
        acc = __builtin_amdgcn_wmma_f32_16x16x32_bf16(false, af, false, bfm,
                                                      (short)0, acc, false, false);
    }
    const int nc = tn*TILE + wn*16 + ml;
#pragma unroll
    for (int r = 0; r < 8; ++r) {
        const int m = tm*TILE + wm*16 + r + 8*half;
        if (m < M && nc < N) {
            float v = acc[r];
            if (bb) v += bb[nc];
            if (act == 1) v = fmaxf(v, 0.f);
            else if (act == 2) v = tanhf(v);
            D[(long)m * N + nc] = v;
        }
    }
}

// =====================================================================
// conv1: 1->32ch, 3x3 s2 p1, 128x128 -> 64x64, ReLU, bf16 out (VALU, K=9).
// Branchless: clamped address + value select.
// =====================================================================
__global__ void conv1_direct(const float* __restrict__ img,
                             const float* __restrict__ w,
                             const float* __restrict__ bias,
                             bf16* __restrict__ out, long nimg)
{
    long idx = (long)blockIdx.x * blockDim.x + threadIdx.x;
    if (idx >= nimg * 32 * 64 * 64) return;
    int ox = idx & 63;
    int oy = (idx >> 6) & 63;
    int oc = (idx >> 12) & 31;
    long b = idx >> 17;
    const float* in = img + b * 128 * 128;
    float acc = bias[oc];
#pragma unroll
    for (int ky = 0; ky < 3; ++ky)
#pragma unroll
        for (int kx = 0; kx < 3; ++kx) {
            int iy = oy*2 - 1 + ky, ix = ox*2 - 1 + kx;
            bool ok = ((unsigned)iy < 128u) && ((unsigned)ix < 128u);
            int iyc = ok ? iy : 0, ixc = ok ? ix : 0;
            float v = in[iyc*128 + ixc];
            acc += w[oc*9 + ky*3 + kx] * (ok ? v : 0.f);
        }
    out[idx] = (bf16)fmaxf(acc, 0.f);
}

// =====================================================================
// conv2 implicit-GEMM WMMA: per image M=1024 px(32x32), N=48, K=288
// =====================================================================
__global__ __launch_bounds__(32) void conv2_wmma(const bf16* __restrict__ c1,
                                                 const float* __restrict__ w,
                                                 const float* __restrict__ bias,
                                                 bf16* __restrict__ c2)
{
    const int img  = blockIdx.y;
    const int tn   = blockIdx.x % 3;
    const int tm   = blockIdx.x / 3;
    const int lane = threadIdx.x & 31;
    const int half = lane >> 4;
    const int ml   = lane & 15;
    const int p    = tm*16 + ml;
    const int oy   = p >> 5, ox = p & 31;
    const bf16* in = c1 + (long)img * 32 * 64 * 64;

    v8f acc = {};
    for (int k0 = 0; k0 < 288; k0 += 32) {
        v16bf av, bv;
#pragma unroll
        for (int j = 0; j < 8; ++j) {
            const int kb = (j < 4) ? (2*j + 8*half) : (16 + 2*(j-4) + 8*half);
#pragma unroll
            for (int tt = 0; tt < 2; ++tt) {
                const int k  = k0 + kb + tt;
                const int ic = k / 9, r = k % 9;
                const int ky = r / 3, kx = r % 3;
                const int iy = oy*2 - 1 + ky, ix = ox*2 - 1 + kx;
                bool ok = ((unsigned)iy < 64u) && ((unsigned)ix < 64u);
                int iyc = ok ? iy : 0, ixc = ok ? ix : 0;
                bf16 v = in[((long)ic*64 + iyc)*64 + ixc];
                av[2*j + tt] = ok ? v : (bf16)0.f;
            }
        }
#pragma unroll
        for (int e = 0; e < 16; ++e) {
            const int kr = k0 + ((e < 8) ? (8*half + e) : (16 + 8*half + e - 8));
            bv[e] = (bf16)w[(tn*16 + ml)*288 + kr];
        }
        acc = __builtin_amdgcn_wmma_f32_16x16x32_bf16(false, av, false, bv,
                                                      (short)0, acc, false, false);
    }
    const int oc = tn*16 + ml;
#pragma unroll
    for (int r = 0; r < 8; ++r) {
        const int m  = tm*16 + r + 8*half;
        const int yy = m >> 5, xx = m & 31;
        float v = fmaxf(acc[r] + bias[oc], 0.f);
        c2[(((long)img*48 + oc)*32 + yy)*32 + xx] = (bf16)v;
    }
}

// maxpool 2x2: [B,48,32,32] -> [B,48,16,16]
__global__ void maxpool2(const bf16* __restrict__ in, bf16* __restrict__ out)
{
    long idx = (long)blockIdx.x * blockDim.x + threadIdx.x;
    if (idx >= (long)NIMG * 48 * 16 * 16) return;
    int ox = idx & 15; long t = idx >> 4;
    int oy = (int)(t & 15); t >>= 4;
    int c  = (int)(t % 48); long b = t / 48;
    const bf16* p = in + (((b*48 + c)*32 + oy*2)*32 + ox*2);
    float v0 = (float)p[0],  v1 = (float)p[1];
    float v2 = (float)p[32], v3 = (float)p[33];
    out[idx] = (bf16)fmaxf(fmaxf(v0, v1), fmaxf(v2, v3));
}

// =====================================================================
// conv3 implicit-GEMM WMMA: per image M=64 px(8x8), N=64, K=432.
// fp32 out in torch-flatten order (oc*64 + oy*8 + ox), ReLU.
// =====================================================================
__global__ __launch_bounds__(32) void conv3_wmma(const bf16* __restrict__ pl,
                                                 const float* __restrict__ w,
                                                 const float* __restrict__ bias,
                                                 float* __restrict__ c3)
{
    const int img  = blockIdx.y;
    const int tn   = blockIdx.x % 4;
    const int tm   = blockIdx.x / 4;
    const int lane = threadIdx.x & 31;
    const int half = lane >> 4;
    const int ml   = lane & 15;
    const int p    = tm*16 + ml;
    const int oy   = p >> 3, ox = p & 7;
    const bf16* in = pl + (long)img * 48 * 16 * 16;

    v8f acc = {};
    for (int k0 = 0; k0 < 432; k0 += 32) {
        v16bf av, bv;
#pragma unroll
        for (int j = 0; j < 8; ++j) {
            const int kb = (j < 4) ? (2*j + 8*half) : (16 + 2*(j-4) + 8*half);
#pragma unroll
            for (int tt = 0; tt < 2; ++tt) {
                const int k  = k0 + kb + tt;
                const int kc = (k < 432) ? k : 431;
                const int ic = kc / 9, r = kc % 9;
                const int ky = r / 3, kx = r % 3;
                const int iy = oy*2 - 1 + ky, ix = ox*2 - 1 + kx;
                bool ok = (k < 432) && ((unsigned)iy < 16u) && ((unsigned)ix < 16u);
                int iyc = ((unsigned)iy < 16u) ? iy : 0;
                int ixc = ((unsigned)ix < 16u) ? ix : 0;
                bf16 v = in[((long)ic*16 + iyc)*16 + ixc];
                av[2*j + tt] = ok ? v : (bf16)0.f;
            }
        }
#pragma unroll
        for (int e = 0; e < 16; ++e) {
            const int kr = k0 + ((e < 8) ? (8*half + e) : (16 + 8*half + e - 8));
            const int krc = (kr < 432) ? kr : 431;
            float wv_ = w[(tn*16 + ml)*432 + krc];
            bv[e] = (bf16)((kr < 432) ? wv_ : 0.f);
        }
        acc = __builtin_amdgcn_wmma_f32_16x16x32_bf16(false, av, false, bv,
                                                      (short)0, acc, false, false);
    }
    const int oc = tn*16 + ml;
#pragma unroll
    for (int r = 0; r < 8; ++r) {
        const int m = tm*16 + r + 8*half;
        c3[(long)img*4096 + oc*64 + m] = fmaxf(acc[r] + bias[oc], 0.f);
    }
}

// ---- small elementwise / reduction kernels ----
__global__ void concat2(const float* __restrict__ a, int ca,
                        const float* __restrict__ b, int cb,
                        float* __restrict__ out, int rows)
{
    long idx = (long)blockIdx.x * blockDim.x + threadIdx.x;
    const int cols = ca + cb;
    if (idx >= (long)rows * cols) return;
    int row = (int)(idx / cols), col = (int)(idx % cols);
    out[idx] = (col < ca) ? a[(long)row*ca + col] : b[(long)row*cb + (col - ca)];
}

__global__ void diag_kernel(const float* __restrict__ x, float* __restrict__ out,
                            int rows, int d, float scale)
{
    int r = blockIdx.x * blockDim.x + threadIdx.x;
    if (r >= rows) return;
    float s = 0.f;
    for (int i = 0; i < d; ++i) { float v = x[(long)r*d + i] * scale; s += v*v; }
    out[r] = 0.5f * s;
}

__global__ void rowmax_kernel(const float* __restrict__ x, float* __restrict__ out,
                              int rows, int cols)
{
    int r = blockIdx.x * blockDim.x + threadIdx.x;
    if (r >= rows) return;
    float m = -1e30f;
    for (int i = 0; i < cols; ++i) m = fmaxf(m, x[(long)r*cols + i]);
    out[r] = m;
}

__global__ void groupmax32_kernel(const float* __restrict__ rmax,
                                  float* __restrict__ gmax, int groups)
{
    int g = blockIdx.x * blockDim.x + threadIdx.x;
    if (g >= groups) return;
    float m = -1e30f;
    for (int i = 0; i < 32; ++i) m = fmaxf(m, rmax[g*32 + i]);
    gmax[g] = m;
}

__global__ void perf_epilogue(float* __restrict__ xd, const float* __restrict__ dg,
                              const float* __restrict__ mx, int mdiv,
                              int rows, int cols, float ratio)
{
    long idx = (long)blockIdx.x * blockDim.x + threadIdx.x;
    if (idx >= (long)rows * cols) return;
    long r = idx / cols;
    xd[idx] = ratio * (expf(xd[idx] - dg[r] - mx[r / mdiv]) + 1e-4f);
}

__global__ void ksum_kernel(const float* __restrict__ kp, float* __restrict__ ks)
{
    long idx = (long)blockIdx.x * blockDim.x + threadIdx.x;
    if (idx >= (long)NHEADS * TT * NFEAT) return;
    int g = (int)(idx / NFEAT), f = (int)(idx % NFEAT);
    float s = 0.f;
    for (int n = 0; n < 32; ++n) s += kp[((long)g*32 + n)*NFEAT + f];
    ks[idx] = s;
}

__global__ void dinv_kernel(const float* __restrict__ qp, const float* __restrict__ ks,
                            float* __restrict__ di)
{
    int r = blockIdx.x * blockDim.x + threadIdx.x;
    if (r >= NHEADS * TT * NQQ) return;
    int g = r >> 5;
    float s = 0.f;
    for (int f = 0; f < NFEAT; ++f)
        s += qp[(long)r*NFEAT + f] * ks[(long)g*NFEAT + f];
    di[r] = 1.f / s;
}

// outs stored [h][b][n][e]; emit [b*NQ+n][e*8+h] * dinv
__global__ void permute_outs(const float* __restrict__ outs, const float* __restrict__ di,
                             float* __restrict__ op)
{
    long idx = (long)blockIdx.x * blockDim.x + threadIdx.x;
    if (idx >= (long)TT * NQQ * NHEADS * DW) return;
    int f   = (int)(idx % (NHEADS * DW));
    int row = (int)(idx / (NHEADS * DW));
    int e = f >> 3, h = f & 7;
    int b = row >> 5, n = row & 31;
    int g = h * TT + b;
    op[idx] = outs[((long)g*32 + n)*DW + e] * di[g*32 + n];
}

__global__ void set_kl(float* __restrict__ out)
{
    if (threadIdx.x == 0) out[TT * NQQ * 2] = 0.f;
}

// =====================================================================
// host-side orchestration
// =====================================================================
extern "C" void kernel_launch(void* const* d_in, const int* in_sizes, int n_in,
                              void* d_out, int out_size, void* d_ws, size_t ws_size,
                              hipStream_t stream)
{
    const float* trainImg = (const float*)d_in[0];
    const float* labels   = (const float*)d_in[1];
    const float* testImg  = (const float*)d_in[2];
    const float* c1w = (const float*)d_in[3];   const float* c1b = (const float*)d_in[4];
    const float* c2w = (const float*)d_in[5];   const float* c2b = (const float*)d_in[6];
    const float* c3w = (const float*)d_in[7];   const float* c3b = (const float*)d_in[8];
    const float* elw = (const float*)d_in[9];   const float* elb = (const float*)d_in[10];
    const float* tyw = (const float*)d_in[11];  const float* tyb = (const float*)d_in[12];
    const float* ew1 = (const float*)d_in[13];  const float* eb1 = (const float*)d_in[14];
    const float* ew2 = (const float*)d_in[15];  const float* eb2 = (const float*)d_in[16];
    const float* ew3 = (const float*)d_in[17];  const float* eb3 = (const float*)d_in[18];
    const float* wq  = (const float*)d_in[19];  const float* wqb = (const float*)d_in[20];
    const float* wk  = (const float*)d_in[21];  const float* wkb = (const float*)d_in[22];
    const float* wv  = (const float*)d_in[23];  const float* wvb = (const float*)d_in[24];
    const float* wow = (const float*)d_in[25];  const float* wob = (const float*)d_in[26];
    const float* rzw = (const float*)d_in[27];  const float* rzb = (const float*)d_in[28];
    const float* dw1 = (const float*)d_in[29];  const float* db1 = (const float*)d_in[30];
    const float* dw2 = (const float*)d_in[31];  const float* db2 = (const float*)d_in[32];
    const float* dw3 = (const float*)d_in[33];  const float* db3 = (const float*)d_in[34];
    const float* proj = (const float*)d_in[35];
    float* outp = (float*)d_out;

    // ---- carve workspace ----
    char* wsp = (char*)d_ws;
    auto carve = [&](size_t bytes) -> void* {
        void* p = (void*)wsp;
        wsp += (bytes + 255) & ~(size_t)255;
        return p;
    };
    // conv pipeline
    bf16*  c1v = (bf16*) carve((size_t)NIMG*32*64*64*2);
    bf16*  c2v = (bf16*) carve((size_t)NIMG*48*32*32*2);
    bf16*  plv = (bf16*) carve((size_t)NIMG*48*16*16*2);
    float* c3v = (float*)carve((size_t)NIMG*4096*4);
    // fp32 activations
    float* xenc  = (float*)carve((size_t)NIMG*DW*4);
    float* ly    = (float*)carve((size_t)512*64*4);
    float* xcat  = (float*)carve((size_t)512*320*4);
    float* h1    = (float*)carve((size_t)512*DR*4);
    float* h2    = (float*)carve((size_t)512*DR*4);
    float* rsv   = (float*)carve((size_t)512*DR*4);
    float* qall  = (float*)carve((size_t)NHEADS*512*DW*4);
    float* kall  = (float*)carve((size_t)NHEADS*512*DW*4);
    float* vall  = (float*)carve((size_t)NHEADS*512*DW*4);
    float* xdq   = (float*)carve((size_t)4096*NFEAT*4);
    float* xdk   = (float*)carve((size_t)4096*NFEAT*4);
    float* diagq = (float*)carve(4096*4);
    float* diagk = (float*)carve(4096*4);
    float* rmaxq = (float*)carve(4096*4);
    float* rmaxk = (float*)carve(4096*4);
    float* gmaxk = (float*)carve(128*4);
    float* ksum  = (float*)carve((size_t)128*NFEAT*4);
    float* dinv  = (float*)carve(4096*4);
    float* ctx   = (float*)carve((size_t)128*NFEAT*DW*4);
    float* outs  = (float*)carve((size_t)NHEADS*512*DW*4);
    float* outsp = (float*)carve((size_t)512*NHEADS*DW*4);
    float* rep   = (float*)carve((size_t)512*DW*4);
    float* zz    = (float*)carve((size_t)512*DZ*4);
    float* xz    = (float*)carve((size_t)512*(DW+DZ)*4);
    float* dh1   = (float*)carve((size_t)512*100*4);
    float* dh2   = (float*)carve((size_t)512*100*4);
    // packed bf16 operands
    bf16* pAenc = (bf16*)carve((size_t)1024*4096*2);
    bf16* pBenc = (bf16*)carve((size_t)4096*256*2);
    bf16* pAty  = (bf16*)carve((size_t)512*32*2);
    bf16* pBty  = (bf16*)carve((size_t)32*64*2);
    bf16* pAe1  = (bf16*)carve((size_t)512*320*2);
    bf16* pBe1  = (bf16*)carve((size_t)320*256*2);
    bf16* pAe2  = (bf16*)carve((size_t)512*256*2);
    bf16* pBe2  = (bf16*)carve((size_t)256*256*2);
    bf16* pAe3  = (bf16*)carve((size_t)512*256*2);
    bf16* pBe3  = (bf16*)carve((size_t)256*256*2);
    bf16* pAq   = (bf16*)carve((size_t)512*256*2);
    bf16* pAk   = (bf16*)carve((size_t)512*256*2);
    bf16* pAv   = (bf16*)carve((size_t)512*256*2);
    bf16* pBwq  = (bf16*)carve((size_t)8*256*256*2);
    bf16* pBwk  = (bf16*)carve((size_t)8*256*256*2);
    bf16* pBwv  = (bf16*)carve((size_t)8*256*256*2);
    bf16* pAxq  = (bf16*)carve((size_t)4096*256*2);
    bf16* pAxk  = (bf16*)carve((size_t)4096*256*2);
    bf16* pBpj  = (bf16*)carve((size_t)256*1440*2);
    bf16* pAkp  = (bf16*)carve((size_t)128*1440*32*2);
    bf16* pBv   = (bf16*)carve((size_t)128*32*256*2);
    bf16* pAqp  = (bf16*)carve((size_t)128*32*1440*2);
    bf16* pBctx = (bf16*)carve((size_t)128*1440*256*2);
    bf16* pAop  = (bf16*)carve((size_t)512*2048*2);
    bf16* pBwo  = (bf16*)carve((size_t)2048*256*2);
    bf16* pArep = (bf16*)carve((size_t)512*256*2);
    bf16* pBrz  = (bf16*)carve((size_t)256*128*2);
    bf16* pAxz  = (bf16*)carve((size_t)512*384*2);
    bf16* pBd1  = (bf16*)carve((size_t)384*128*2);
    bf16* pAd1  = (bf16*)carve((size_t)512*128*2);
    bf16* pBd2  = (bf16*)carve((size_t)128*128*2);
    bf16* pAd2  = (bf16*)carve((size_t)512*128*2);
    bf16* pBd3  = (bf16*)carve((size_t)128*32*2);

    auto packA = [&](const float* in, bf16* out, int M, int K, int Mpad, int Kpad,
                     long sAm, long sAk, long bsIn, int nb, float ascale) {
        long tot = (long)Mpad * Kpad * nb;
        pack_a_kernel<<<(unsigned)((tot + 255)/256), 256, 0, stream>>>(
            in, out, M, K, Mpad, Kpad, sAm, sAk, bsIn, nb, ascale);
    };
    auto packB = [&](const float* in, bf16* out, int N, int K, int Npad, int Kpad,
                     long sBn, long sBk, long bsIn, int nb) {
        long tot = (long)Kpad * Npad * nb;
        pack_b_kernel<<<(unsigned)((tot + 255)/256), 256, 0, stream>>>(
            in, out, N, K, Npad, Kpad, sBn, sBk, bsIn, nb);
    };
    auto gemm = [&](const bf16* Ap, const bf16* Bp, const float* bias, float* D,
                    int M, int N, int Kpad, int Mpad, int Npad, int nb,
                    long bsA, long bsB, long bsBias, long bsD, int act) {
        dim3 g((unsigned)((Mpad/32) * (Npad/32)), (unsigned)nb);
        gemm_pk_wmma<<<g, 128, 0, stream>>>(Ap, Bp, bias, D, M, N, Kpad, Npad,
                                            bsA, bsB, bsBias, bsD, act);
    };
    const float ratio = 1.0f / sqrtf((float)NFEAT);

    // ---- encoder CNN ----
    {
        long per = (long)512*32*64*64;
        conv1_direct<<<(unsigned)((per + 255)/256), 256, 0, stream>>>(trainImg, c1w, c1b, c1v, 512);
        conv1_direct<<<(unsigned)((per + 255)/256), 256, 0, stream>>>(testImg,  c1w, c1b, c1v + per, 512);
    }
    conv2_wmma<<<dim3(192, NIMG), 32, 0, stream>>>(c1v, c2w, c2b, c2v);
    {
        long tot = (long)NIMG*48*16*16;
        maxpool2<<<(unsigned)((tot + 255)/256), 256, 0, stream>>>(c2v, plv);
    }
    conv3_wmma<<<dim3(16, NIMG), 32, 0, stream>>>(plv, c3w, c3b, c3v);

    // enc linear: [1024x4096] @ [256x4096]^T
    packA(c3v, pAenc, 1024, 4096, 1024, 4096, 4096, 1, 0, 1, 1.f);
    packB(elw, pBenc, 256, 4096, 256, 4096, 4096, 1, 0, 1);
    gemm(pAenc, pBenc, elb, xenc, 1024, 256, 4096, 1024, 256, 1, 0,0,0,0, 0);

    // label proj
    packA(labels, pAty, 512, 2, 512, 32, 2, 1, 0, 1, 1.f);
    packB(tyw, pBty, 64, 2, 64, 32, 2, 1, 0, 1);
    gemm(pAty, pBty, tyb, ly, 512, 64, 32, 512, 64, 1, 0,0,0,0, 0);

    concat2<<<(unsigned)((512L*320 + 255)/256), 256, 0, stream>>>(xenc, DW, ly, 64, xcat, 512);

    // er MLP
    packA(xcat, pAe1, 512, 320, 512, 320, 320, 1, 0, 1, 1.f);
    packB(ew1, pBe1, 256, 320, 256, 320, 320, 1, 0, 1);
    gemm(pAe1, pBe1, eb1, h1, 512, 256, 320, 512, 256, 1, 0,0,0,0, 1);
    packA(h1, pAe2, 512, 256, 512, 256, 256, 1, 0, 1, 1.f);
    packB(ew2, pBe2, 256, 256, 256, 256, 256, 1, 0, 1);
    gemm(pAe2, pBe2, eb2, h2, 512, 256, 256, 512, 256, 1, 0,0,0,0, 1);
    packA(h2, pAe3, 512, 256, 512, 256, 256, 1, 0, 1, 1.f);
    packB(ew3, pBe3, 256, 256, 256, 256, 256, 1, 0, 1);
    gemm(pAe3, pBe3, eb3, rsv, 512, 256, 256, 512, 256, 1, 0,0,0,0, 0);

    // q/k/v per-head projections (batched over 8 heads), layout [h][512][256]
    packA(xenc + 512L*DW, pAq, 512, 256, 512, 256, 256, 1, 0, 1, 1.f);
    packA(xenc,           pAk, 512, 256, 512, 256, 256, 1, 0, 1, 1.f);
    packA(rsv,            pAv, 512, 256, 512, 256, 256, 1, 0, 1, 1.f);
    packB(wq, pBwq, 256, 256, 256, 256, 256, 1, 65536, 8);
    packB(wk, pBwk, 256, 256, 256, 256, 256, 1, 65536, 8);
    packB(wv, pBwv, 256, 256, 256, 256, 256, 1, 65536, 8);
    gemm(pAq, pBwq, wqb, qall, 512, 256, 256, 512, 256, 8, 0, 65536, 256, 512L*256, 0);
    gemm(pAk, pBwk, wkb, kall, 512, 256, 256, 512, 256, 8, 0, 65536, 256, 512L*256, 0);
    gemm(pAv, pBwv, wvb, vall, 512, 256, 256, 512, 256, 8, 0, 65536, 256, 512L*256, 0);

    // performer features: xd = (x*dn) @ proj^T, dn = 256^-0.25 = 0.25 (folded into A pack)
    packA(qall, pAxq, 4096, 256, 4096, 256, 256, 1, 0, 1, 0.25f);
    packA(kall, pAxk, 4096, 256, 4096, 256, 256, 1, 0, 1, 0.25f);
    packB(proj, pBpj, 1419, 256, 1440, 256, 256, 1, 0, 1);
    gemm(pAxq, pBpj, nullptr, xdq, 4096, 1419, 256, 4096, 1440, 1, 0,0,0,0, 0);
    gemm(pAxk, pBpj, nullptr, xdk, 4096, 1419, 256, 4096, 1440, 1, 0,0,0,0, 0);

    diag_kernel<<<16, 256, 0, stream>>>(qall, diagq, 4096, DW, 0.25f);
    diag_kernel<<<16, 256, 0, stream>>>(kall, diagk, 4096, DW, 0.25f);
    rowmax_kernel<<<16, 256, 0, stream>>>(xdq, rmaxq, 4096, NFEAT);
    rowmax_kernel<<<16, 256, 0, stream>>>(xdk, rmaxk, 4096, NFEAT);
    groupmax32_kernel<<<1, 128, 0, stream>>>(rmaxk, gmaxk, 128);
    {
        long tot = 4096L * NFEAT;
        perf_epilogue<<<(unsigned)((tot + 255)/256), 256, 0, stream>>>(xdq, diagq, rmaxq, 1,  4096, NFEAT, ratio);
        perf_epilogue<<<(unsigned)((tot + 255)/256), 256, 0, stream>>>(xdk, diagk, gmaxk, 32, 4096, NFEAT, ratio);
    }
    {
        long tot = 128L * NFEAT;
        ksum_kernel<<<(unsigned)((tot + 255)/256), 256, 0, stream>>>(xdk, ksum);
    }
    dinv_kernel<<<16, 256, 0, stream>>>(xdq, ksum, dinv);

    // ctx[g][f][e] = sum_n kp[g,n,f] * v[g,n,e]  (g = h*16+b, 128 batches)
    packA(xdk, pAkp, 1419, 32, 1440, 32, 1, NFEAT, 32L*NFEAT, 128, 1.f);
    packB(vall, pBv, 256, 32, 256, 32, 1, 256, 32L*256, 128);
    gemm(pAkp, pBv, nullptr, ctx, 1419, 256, 32, 1440, 256, 128,
         1440L*32, 32L*256, 0, (long)NFEAT*DW, 0);

    // outs[g][n][e] = sum_f qp[g,n,f] * ctx[g,f,e]
    packA(xdq, pAqp, 32, 1419, 32, 1440, NFEAT, 1, 32L*NFEAT, 128, 1.f);
    packB(ctx, pBctx, 256, 1419, 256, 1440, 1, 256, (long)NFEAT*DW, 128);
    gemm(pAqp, pBctx, nullptr, outs, 32, 256, 1440, 32, 256, 128,
         32L*1440, 1440L*256, 0, 32L*256, 0);

    {
        long tot = 512L * NHEADS * DW;
        permute_outs<<<(unsigned)((tot + 255)/256), 256, 0, stream>>>(outs, dinv, outsp);
    }

    // output head
    packA(outsp, pAop, 512, 2048, 512, 2048, 2048, 1, 0, 1, 1.f);
    packB(wow, pBwo, 256, 2048, 256, 2048, 2048, 1, 0, 1);
    gemm(pAop, pBwo, wob, rep, 512, 256, 2048, 512, 256, 1, 0,0,0,0, 0);
    packA(rep, pArep, 512, 256, 512, 256, 256, 1, 0, 1, 1.f);
    packB(rzw, pBrz, 128, 256, 128, 256, 256, 1, 0, 1);
    gemm(pArep, pBrz, rzb, zz, 512, 128, 256, 512, 128, 1, 0,0,0,0, 0);

    concat2<<<(unsigned)((512L*(DW+DZ) + 255)/256), 256, 0, stream>>>(xenc + 512L*DW, DW, zz, DZ, xz, 512);

    packA(xz, pAxz, 512, 384, 512, 384, 384, 1, 0, 1, 1.f);
    packB(dw1, pBd1, 100, 384, 128, 384, 384, 1, 0, 1);
    gemm(pAxz, pBd1, db1, dh1, 512, 100, 384, 512, 128, 1, 0,0,0,0, 1);
    packA(dh1, pAd1, 512, 100, 512, 128, 100, 1, 0, 1, 1.f);
    packB(dw2, pBd2, 100, 100, 128, 128, 100, 1, 0, 1);
    gemm(pAd1, pBd2, db2, dh2, 512, 100, 128, 512, 128, 1, 0,0,0,0, 1);
    packA(dh2, pAd2, 512, 100, 512, 128, 100, 1, 0, 1, 1.f);
    packB(dw3, pBd3, 2, 100, 32, 128, 100, 1, 0, 1);
    gemm(pAd2, pBd3, db3, outp, 512, 2, 128, 512, 32, 1, 0,0,0,0, 2);

    set_kl<<<1, 32, 0, stream>>>(outp);
    (void)in_sizes; (void)n_in; (void)out_size; (void)ws_size;
}
